// GNNModel_53979148976761
// MI455X (gfx1250) — matrix operand order repaired and checked by reading, not compiled, not run on hardware
//
#include <hip/hip_runtime.h>
#include <hip/hip_bf16.h>
#include <math.h>

// ---------------------------------------------------------------------------
// Types for CDNA5 WMMA (wave32)
// ---------------------------------------------------------------------------
typedef __bf16 bf16_t;
typedef __attribute__((ext_vector_type(16))) __bf16 bf16x16;
typedef __attribute__((ext_vector_type(8)))  float  v8f;

#define NROWS          8192
#define ROWS_PER_BLOCK 128   // 8 waves * 16 rows
#define TPB            256
#define KSPLIT         5     // K-partitions for the big GEMM (405 ksteps / 5 = 81)
#define JSPLIT         16    // j-partitions for the kNN search

// ---------------------------------------------------------------------------
// fp32 -> bf16 weight conversion (once per launch, for the big GEMM)
// ---------------------------------------------------------------------------
__global__ void convert_bf16_kernel(const float* __restrict__ src,
                                    bf16_t* __restrict__ dst, int n) {
  int i = blockIdx.x * blockDim.x + threadIdx.x;
  if (i < n) dst[i] = (bf16_t)src[i];
}

// ---------------------------------------------------------------------------
// Big GEMM, K-split: partial[p][8192,64] = x[:, k0:k1] @ Wbf[:, k0:k1]^T
// grid = (64, KSPLIT) -> 2560 waves. Hoisted constant-stride pointers; plain
// loads at loop top + unroll-2 so the backend software-pipelines without
// exceeding the register budget (~100 VGPRs, no spills).
// ---------------------------------------------------------------------------
__global__ void gemm_big_part_kernel(const float* __restrict__ x,
                                     const bf16_t* __restrict__ wbf,
                                     float* __restrict__ partial,
                                     int K, int kstepsPer) {
  const int lane = threadIdx.x & 31;
  const int wave = threadIdx.x >> 5;
  const int col  = lane & 15;
  const int half = lane >> 4;
  const int m    = blockIdx.x * ROWS_PER_BLOCK + wave * 16 + col;
  const int p    = blockIdx.y;
  const int k0   = p * kstepsPer * 32;
  const int k1   = min(K, k0 + kstepsPer * 32);
  const int nsteps = (k1 - k0) / 32;
  float* outp = partial + (size_t)p * NROWS * 64;

  // Hoisted per-iteration-advancing pointers (constant 32-element step).
  const float4* pa0 = (const float4*)(x + (size_t)m * K + k0 + half * 16);
  const float4* pa1 = (const float4*)(x + (size_t)m * K + k0 + 8 + half * 16);
  const bf16_t* pb0 = wbf + (size_t)(0  + col) * K + k0 + half * 16;
  const bf16_t* pb1 = wbf + (size_t)(16 + col) * K + k0 + half * 16;
  const bf16_t* pb2 = wbf + (size_t)(32 + col) * K + k0 + half * 16;
  const bf16_t* pb3 = wbf + (size_t)(48 + col) * K + k0 + half * 16;

  v8f acc[4];
#pragma unroll
  for (int nt = 0; nt < 4; ++nt)
#pragma unroll
    for (int r = 0; r < 8; ++r) acc[nt][r] = 0.0f;

#pragma unroll 2
  for (int s = 0; s < nsteps; ++s) {
    float4 f0 = pa0[0], f1 = pa0[1], f2 = pa1[0], f3 = pa1[1];
    bf16x16 b0 = *(const bf16x16*)pb0;
    bf16x16 b1 = *(const bf16x16*)pb1;
    bf16x16 b2 = *(const bf16x16*)pb2;
    bf16x16 b3 = *(const bf16x16*)pb3;
    pa0 += 8; pa1 += 8;                          // 32 floats
    pb0 += 32; pb1 += 32; pb2 += 32; pb3 += 32;  // 32 bf16

    float fa[16] = {f0.x, f0.y, f0.z, f0.w, f1.x, f1.y, f1.z, f1.w,
                    f2.x, f2.y, f2.z, f2.w, f3.x, f3.y, f3.z, f3.w};
    bf16x16 a;
#pragma unroll
    for (int i = 0; i < 16; ++i) a[i] = (bf16_t)fa[i];

    acc[0] = __builtin_amdgcn_wmma_f32_16x16x32_bf16(false, a, false, b0, (short)0, acc[0], false, false);
    acc[1] = __builtin_amdgcn_wmma_f32_16x16x32_bf16(false, a, false, b1, (short)0, acc[1], false, false);
    acc[2] = __builtin_amdgcn_wmma_f32_16x16x32_bf16(false, a, false, b2, (short)0, acc[2], false, false);
    acc[3] = __builtin_amdgcn_wmma_f32_16x16x32_bf16(false, a, false, b3, (short)0, acc[3], false, false);
  }

#pragma unroll
  for (int nt = 0; nt < 4; ++nt) {
    int n = nt * 16 + col;
#pragma unroll
    for (int r = 0; r < 8; ++r) {
      int mm = blockIdx.x * ROWS_PER_BLOCK + wave * 16 + half * 8 + r;
      outp[(size_t)mm * 64 + n] = acc[nt][r];
    }
  }
}

// out[idx] = sum_p partial[p][idx] + bias[idx%64]   (fixed order -> deterministic)
__global__ void reduce_parts_kernel(const float* __restrict__ partials,
                                    const float* __restrict__ bias,
                                    float* __restrict__ out, int total) {
  int idx = blockIdx.x * blockDim.x + threadIdx.x;
  if (idx >= total) return;
  float v = bias[idx & 63];
#pragma unroll
  for (int p = 0; p < KSPLIT; ++p) v += partials[(size_t)p * total + idx];
  out[idx] = v;
}

// ---------------------------------------------------------------------------
// Generic WMMA GEMM: out[8192,N] = x[8192,K(ldx)] @ w[N,K(ldw)]^T + b
// K, N arbitrary (zero-padded into bf16 fragments). NT = ceil(N/16).
// ---------------------------------------------------------------------------
template <int NT>
__global__ void gemm_wmma_kernel(const float* __restrict__ x, int ldx,
                                 const float* __restrict__ w, int ldw,
                                 const float* __restrict__ bias,
                                 float* __restrict__ out, int ldo,
                                 int K, int N, int relu) {
  const int lane = threadIdx.x & 31;
  const int wave = threadIdx.x >> 5;
  const int col  = lane & 15;
  const int half = lane >> 4;
  const int m    = blockIdx.x * ROWS_PER_BLOCK + wave * 16 + col;
  const float* xr = x + (size_t)m * ldx;

  v8f acc[NT];
#pragma unroll
  for (int nt = 0; nt < NT; ++nt)
#pragma unroll
    for (int r = 0; r < 8; ++r) acc[nt][r] = 0.0f;

  for (int kt = 0; kt < K; kt += 32) {
    bf16x16 a;
#pragma unroll
    for (int i = 0; i < 8; ++i) {
      int k0 = kt + half * 16 + i;
      int k1 = kt + 8 + half * 16 + i;
      a[i]     = (bf16_t)((k0 < K) ? xr[k0] : 0.0f);
      a[i + 8] = (bf16_t)((k1 < K) ? xr[k1] : 0.0f);
    }
#pragma unroll
    for (int nt = 0; nt < NT; ++nt) {
      int n = nt * 16 + col;
      bool nok = (n < N);
      const float* wr = w + (size_t)(nok ? n : 0) * ldw;
      bf16x16 bm;
#pragma unroll
      for (int i = 0; i < 16; ++i) {
        int kk = kt + half * 16 + i;
        bm[i] = (bf16_t)((nok && kk < K) ? wr[kk] : 0.0f);
      }
      acc[nt] = __builtin_amdgcn_wmma_f32_16x16x32_bf16(
          false, a, false, bm, (short)0, acc[nt], false, false);
    }
  }

#pragma unroll
  for (int nt = 0; nt < NT; ++nt) {
    int n = nt * 16 + col;
    if (n < N) {
      float bv = bias[n];
#pragma unroll
      for (int r = 0; r < 8; ++r) {
        int mm = blockIdx.x * ROWS_PER_BLOCK + wave * 16 + half * 8 + r;
        float v = acc[nt][r] + bv;
        if (relu) v = fmaxf(v, 0.0f);
        out[(size_t)mm * ldo + n] = v;
      }
    }
  }
}

// ---------------------------------------------------------------------------
// Fused s (3-dim) + h (1-dim) projections: one pass over each activation row.
// ---------------------------------------------------------------------------
__global__ void proj_sh_kernel(const float* __restrict__ x,
                               const float* __restrict__ ws, const float* __restrict__ bs,
                               const float* __restrict__ wh, const float* __restrict__ bh,
                               float* __restrict__ sout, float* __restrict__ hout, int K) {
  int n = blockIdx.x * blockDim.x + threadIdx.x;
  const float* xr = x + (size_t)n * K;
  const float* w0 = ws;
  const float* w1 = ws + K;
  const float* w2 = ws + 2 * K;
  float a0 = bs[0], a1 = bs[1], a2 = bs[2], a3 = bh[0];
  for (int k = 0; k < K; ++k) {
    float xv = xr[k];
    a0 = fmaf(xv, w0[k], a0);
    a1 = fmaf(xv, w1[k], a1);
    a2 = fmaf(xv, w2[k], a2);
    a3 = fmaf(xv, wh[k], a3);
  }
  sout[n * 3 + 0] = a0;
  sout[n * 3 + 1] = a1;
  sout[n * 3 + 2] = a2;
  hout[n] = a3;
}

// ---------------------------------------------------------------------------
// kNN top-3 (self included), split over JSPLIT j-partitions for parallelism.
// ---------------------------------------------------------------------------
__device__ __forceinline__ void top3_insert(float d, float hv,
                                            float& bd0, float& bd1, float& bd2,
                                            float& bh0, float& bh1, float& bh2) {
  if (d < bd2) {
    if (d < bd1) {
      bd2 = bd1; bh2 = bh1;
      if (d < bd0) { bd1 = bd0; bh1 = bh0; bd0 = d; bh0 = hv; }
      else         { bd1 = d;   bh1 = hv; }
    } else { bd2 = d; bh2 = hv; }
  }
}

__global__ void knn_part_kernel(const float* __restrict__ s,
                                const float* __restrict__ h,
                                float* __restrict__ pd, float* __restrict__ ph, int N) {
  __shared__ float4 tile[TPB];
  int i = blockIdx.x * blockDim.x + threadIdx.x;
  int p = blockIdx.y;
  int jlen = N / JSPLIT;
  int jbeg = p * jlen;
  float s0 = s[i * 3 + 0], s1 = s[i * 3 + 1], s2 = s[i * 3 + 2];
  float bd0 = 3.4e38f, bd1 = 3.4e38f, bd2 = 3.4e38f;
  float bh0 = 0.f, bh1 = 0.f, bh2 = 0.f;

  for (int j0 = jbeg; j0 < jbeg + jlen; j0 += TPB) {
    int j = j0 + threadIdx.x;
    __syncthreads();
    tile[threadIdx.x] = make_float4(s[j * 3 + 0], s[j * 3 + 1], s[j * 3 + 2], h[j]);
    __syncthreads();
#pragma unroll 4
    for (int jj = 0; jj < TPB; ++jj) {
      float4 tq = tile[jj];
      float dx = s0 - tq.x, dy = s1 - tq.y, dz = s2 - tq.z;
      float d = fmaf(dx, dx, fmaf(dy, dy, dz * dz));
      top3_insert(d, tq.w, bd0, bd1, bd2, bh0, bh1, bh2);
    }
  }
  size_t o = ((size_t)p * N + i) * 3;
  pd[o + 0] = bd0; pd[o + 1] = bd1; pd[o + 2] = bd2;
  ph[o + 0] = bh0; ph[o + 1] = bh1; ph[o + 2] = bh2;
}

__global__ void knn_merge_kernel(const float* __restrict__ pd,
                                 const float* __restrict__ ph,
                                 float* __restrict__ agg, int N) {
  int i = blockIdx.x * blockDim.x + threadIdx.x;
  float bd0 = 3.4e38f, bd1 = 3.4e38f, bd2 = 3.4e38f;
  float bh0 = 0.f, bh1 = 0.f, bh2 = 0.f;
#pragma unroll
  for (int p = 0; p < JSPLIT; ++p) {
    size_t o = ((size_t)p * N + i) * 3;
#pragma unroll
    for (int k = 0; k < 3; ++k)
      top3_insert(pd[o + k], ph[o + k], bd0, bd1, bd2, bh0, bh1, bh2);
  }
  float m0 = bh0 * __expf(-10.f * bd0);
  float m1 = bh1 * __expf(-10.f * bd1);
  float m2 = bh2 * __expf(-10.f * bd2);
  agg[i * 2 + 0] = (m0 + m1 + m2) * (1.0f / 3.0f);
  agg[i * 2 + 1] = fmaxf(m0, fmaxf(m1, m2));
}

// out[r,c] += agg[r,0]*Wo[c,fin] + agg[r,1]*Wo[c,fin+1]  (the concat tail)
__global__ void add_agg_kernel(const float* __restrict__ agg,
                               const float* __restrict__ w,
                               float* __restrict__ out, int N, int ldw, int fin) {
  int idx = blockIdx.x * blockDim.x + threadIdx.x;
  int r = idx / N, c = idx % N;
  if (r >= NROWS) return;
  float a0 = agg[r * 2], a1 = agg[r * 2 + 1];
  out[idx] += a0 * w[(size_t)c * ldw + fin] + a1 * w[(size_t)c * ldw + fin + 1];
}

__global__ void concat3_kernel(const float* __restrict__ x1,
                               const float* __restrict__ x2,
                               const float* __restrict__ x3,
                               float* __restrict__ out) {
  int idx = blockIdx.x * blockDim.x + threadIdx.x;
  int r = idx / 192, c = idx % 192;
  if (r >= NROWS) return;
  float v = (c < 64)  ? x1[r * 64 + c]
          : (c < 128) ? x2[r * 64 + (c - 64)]
                      : x3[r * 64 + (c - 128)];
  out[idx] = v;
}

// BatchNorm1d training stats -> per-channel scale/shift
__global__ void bn_stats_kernel(const float* __restrict__ x,
                                const float* __restrict__ gamma,
                                const float* __restrict__ beta,
                                float* __restrict__ scale,
                                float* __restrict__ shift, int M, int C) {
  __shared__ float ssum[TPB], ssq[TPB];
  int c = blockIdx.x;
  float sum = 0.f, sq = 0.f;
  for (int r = threadIdx.x; r < M; r += TPB) {
    float v = x[(size_t)r * C + c];
    sum += v; sq = fmaf(v, v, sq);
  }
  ssum[threadIdx.x] = sum; ssq[threadIdx.x] = sq;
  __syncthreads();
  for (int st = TPB / 2; st > 0; st >>= 1) {
    if (threadIdx.x < st) {
      ssum[threadIdx.x] += ssum[threadIdx.x + st];
      ssq[threadIdx.x]  += ssq[threadIdx.x + st];
    }
    __syncthreads();
  }
  if (threadIdx.x == 0) {
    float mu  = ssum[0] / M;
    float var = ssq[0] / M - mu * mu;
    float rs  = rsqrtf(var + 1e-5f);
    scale[c] = gamma[c] * rs;
    shift[c] = beta[c] - gamma[c] * rs * mu;
  }
}

__global__ void bn_apply_kernel(float* __restrict__ x,
                                const float* __restrict__ scale,
                                const float* __restrict__ shift,
                                int total, int C) {
  int idx = blockIdx.x * blockDim.x + threadIdx.x;
  if (idx >= total) return;
  int c = idx % C;
  x[idx] = fmaf(x[idx], scale[c], shift[c]);
}

// ---------------------------------------------------------------------------
// Host orchestration
// ---------------------------------------------------------------------------
struct LinP { const float* w; const float* b; };

extern "C" void kernel_launch(void* const* d_in, const int* in_sizes, int n_in,
                              void* d_out, int out_size, void* d_ws, size_t ws_size,
                              hipStream_t stream) {
  (void)in_sizes; (void)n_in; (void)out_size; (void)ws_size;

  int t = 0;
  auto nxt  = [&]() { return (const float*)d_in[t++]; };
  auto take = [&]() { LinP p; p.w = nxt(); p.b = nxt(); return p; };

  const float* X = nxt();  // input_hits [8192, 12960]

  // ---- workspace carve ----
  size_t off = 0;
  auto alloc = [&](size_t bytes) -> void* {
    void* p = (char*)d_ws + off;
    off += (bytes + 255) & ~(size_t)255;
    return p;
  };
  bf16_t* Wbf   = (bf16_t*)alloc((size_t)64 * 12960 * sizeof(bf16_t));
  float* parts  = (float*)alloc((size_t)KSPLIT * NROWS * 64 * sizeof(float));
  float* bufA   = (float*)alloc((size_t)NROWS * 64 * sizeof(float));
  float* bufB   = (float*)alloc((size_t)NROWS * 64 * sizeof(float));
  float* x1     = (float*)alloc((size_t)NROWS * 64 * sizeof(float));
  float* x2     = (float*)alloc((size_t)NROWS * 64 * sizeof(float));
  float* x3     = (float*)alloc((size_t)NROWS * 64 * sizeof(float));
  float* sbuf   = (float*)alloc((size_t)NROWS * 3 * sizeof(float));
  float* hbuf   = (float*)alloc((size_t)NROWS * 1 * sizeof(float));
  float* aggb   = (float*)alloc((size_t)NROWS * 2 * sizeof(float));
  float* pdbuf  = (float*)alloc((size_t)JSPLIT * NROWS * 3 * sizeof(float));
  float* phbuf  = (float*)alloc((size_t)JSPLIT * NROWS * 3 * sizeof(float));
  float* cat    = (float*)alloc((size_t)NROWS * 192 * sizeof(float));
  float* bnsc   = (float*)alloc(64 * sizeof(float));
  float* bnsh   = (float*)alloc(64 * sizeof(float));

  const dim3 gGemm(NROWS / ROWS_PER_BLOCK), bGemm(TPB);

  auto gemm = [&](const float* xp, int ldx, const float* wp, int ldw,
                  const float* bp, float* op, int ldo, int K, int N, int relu) {
    if (N <= 16)
      gemm_wmma_kernel<1><<<gGemm, bGemm, 0, stream>>>(xp, ldx, wp, ldw, bp, op, ldo, K, N, relu);
    else if (N <= 32)
      gemm_wmma_kernel<2><<<gGemm, bGemm, 0, stream>>>(xp, ldx, wp, ldw, bp, op, ldo, K, N, relu);
    else if (N <= 64)
      gemm_wmma_kernel<4><<<gGemm, bGemm, 0, stream>>>(xp, ldx, wp, ldw, bp, op, ldo, K, N, relu);
    else
      gemm_wmma_kernel<12><<<gGemm, bGemm, 0, stream>>>(xp, ldx, wp, ldw, bp, op, ldo, K, N, relu);
  };

  auto run_block = [&](const float* xin, int in_ch, float* xout) {
    LinP d1 = take(), d2 = take(), d3 = take();
    LinP mps[6], mph[6], mpo[6];
    for (int i = 0; i < 6; ++i) { mps[i] = take(); mph[i] = take(); mpo[i] = take(); }
    LinP d4 = take(), d5 = take(), d6 = take();

    if (in_ch > 1024) {
      int nw = 64 * in_ch;
      convert_bf16_kernel<<<(nw + TPB - 1) / TPB, TPB, 0, stream>>>(d1.w, Wbf, nw);
      int ksteps = in_ch / 32;
      int kper = (ksteps + KSPLIT - 1) / KSPLIT;
      gemm_big_part_kernel<<<dim3(NROWS / ROWS_PER_BLOCK, KSPLIT), bGemm, 0, stream>>>(
          xin, Wbf, parts, in_ch, kper);
      reduce_parts_kernel<<<(NROWS * 64) / TPB, TPB, 0, stream>>>(
          parts, d1.b, bufA, NROWS * 64);
    } else {
      gemm(xin, in_ch, d1.w, in_ch, d1.b, bufA, 64, in_ch, 64, 0);
    }
    gemm(bufA, 64, d2.w, 64, d2.b, bufB, 64, 64, 64, 1);
    gemm(bufB, 64, d3.w, 64, d3.b, bufA, 64, 64, 64, 1);

    float* cur = bufA; float* oth = bufB;
    int ch = 64;
    const int fouts[6] = {64, 64, 32, 32, 16, 16};
    for (int i = 0; i < 6; ++i) {
      int fo = fouts[i];
      proj_sh_kernel<<<NROWS / TPB, TPB, 0, stream>>>(
          cur, mps[i].w, mps[i].b, mph[i].w, mph[i].b, sbuf, hbuf, ch);
      knn_part_kernel<<<dim3(NROWS / TPB, JSPLIT), TPB, 0, stream>>>(
          sbuf, hbuf, pdbuf, phbuf, NROWS);
      knn_merge_kernel<<<NROWS / TPB, TPB, 0, stream>>>(pdbuf, phbuf, aggb, NROWS);
      gemm(cur, ch, mpo[i].w, ch + 2, mpo[i].b, oth, fo, ch, fo, 0);
      add_agg_kernel<<<((size_t)NROWS * fo + TPB - 1) / TPB, TPB, 0, stream>>>(
          aggb, mpo[i].w, oth, fo, ch + 2, ch);
      float* tmp = cur; cur = oth; oth = tmp; ch = fo;
    }
    gemm(cur, 16, d4.w, 16, d4.b, oth, 64, 16, 64, 1);
    gemm(oth, 64, d5.w, 64, d5.b, cur, 64, 64, 64, 1);
    gemm(cur, 64, d6.w, 64, d6.b, xout, 64, 64, 64, 1);
  };

  run_block(X, 12960, x1);
  run_block(x1, 64, x2);
  run_block(x2, 64, x3);

  LinP fc1 = take(), fc2 = take(), fc3 = take();
  const float* gamma = nxt();
  const float* beta  = nxt();
  LinP fc4 = take();

  concat3_kernel<<<((size_t)NROWS * 192 + TPB - 1) / TPB, TPB, 0, stream>>>(x1, x2, x3, cat);
  gemm(cat, 192, fc1.w, 192, fc1.b, bufA, 64, 192, 64, 1);
  gemm(bufA, 64, fc2.w, 64, fc2.b, bufB, 64, 64, 64, 1);
  gemm(bufB, 64, fc3.w, 64, fc3.b, bufA, 64, 64, 64, 1);
  bn_stats_kernel<<<64, TPB, 0, stream>>>(bufA, gamma, beta, bnsc, bnsh, NROWS, 64);
  bn_apply_kernel<<<((size_t)NROWS * 64 + TPB - 1) / TPB, TPB, 0, stream>>>(
      bufA, bnsc, bnsh, NROWS * 64, 64);
  gemm(bufA, 64, fc4.w, 64, fc4.b, (float*)d_out, 6, 64, 6, 1);
}
